// CausalAttention_82841329205422
// MI455X (gfx1250) — matrix operand order repaired and checked by reading, not compiled, not run on hardware
//
#include <hip/hip_runtime.h>
#include <hip/hip_bf16.h>

typedef __attribute__((ext_vector_type(16))) _Float16 v16h;
typedef __attribute__((ext_vector_type(8)))  float    v8f;

#define B_   2
#define H_   12
#define T_   2048
#define D_   64
#define DIM_ 768

union V16 { v16h v; uint4 q[2]; _Float16 h[16]; };
union H8  { _Float16 h[8]; unsigned u[4]; uint4 q; };
union F8  { v8f v; float f[8]; };

__device__ __forceinline__ unsigned shfl16u(unsigned x) {
  return (unsigned)__shfl_xor((int)x, 16, 32);
}
__device__ __forceinline__ float shfl16f(float x) {
  return __shfl_xor(x, 16, 32);
}

// ---------------------------------------------------------------------------
// Prep: k -> f16 (same layout); v -> f16 transposed (B,H,D,T); W -> f16 W^T
// ---------------------------------------------------------------------------
__global__ void prep_kernel(const float* __restrict__ k,
                            const float* __restrict__ v,
                            const float* __restrict__ W,
                            _Float16* __restrict__ k16,
                            _Float16* __restrict__ vt16,
                            _Float16* __restrict__ wt16) {
  const size_t NKV = (size_t)B_ * H_ * T_ * D_;
  const size_t NW  = (size_t)DIM_ * DIM_;
  const size_t total = 2 * NKV + NW;
  size_t stride = (size_t)gridDim.x * blockDim.x;
  for (size_t i = (size_t)blockIdx.x * blockDim.x + threadIdx.x; i < total; i += stride) {
    if (i < NKV) {
      k16[i] = (_Float16)k[i];
    } else if (i < 2 * NKV) {
      size_t j = i - NKV;            // vt index: ((bh*64 + d)*T + t)
      size_t t = j % T_;
      size_t r = j / T_;
      size_t d = r % D_;
      size_t bh = r / D_;
      vt16[j] = (_Float16)v[(bh * T_ + t) * D_ + d];
    } else {
      size_t j = i - 2 * NKV;        // wt index: n*DIM + kk  (= W[kk][n])
      size_t kk = j % DIM_;
      size_t n  = j / DIM_;
      wt16[j] = (_Float16)W[kk * DIM_ + n];
    }
  }
}

// ---------------------------------------------------------------------------
// Flash attention: 1 wave = 16 query rows of one (b,h). Computes S^T = K Q^T
// so per-query softmax state is per-lane; y^T = V^T P^T accumulated in 4 C
// tiles; final y stored f16 in (B, T, H*D) layout.
// ---------------------------------------------------------------------------
__global__ __launch_bounds__(128) void attn_kernel(
    const float* __restrict__ q, const _Float16* __restrict__ k16,
    const _Float16* __restrict__ vt16, const float* __restrict__ bias,
    _Float16* __restrict__ y16) {
  const int wave = blockIdx.x * (blockDim.x >> 5) + (threadIdx.x >> 5);
  const int lane = threadIdx.x & 31;
  const int lq = lane & 15;     // query column (and A-row index)
  const int hf = lane >> 4;     // half-wave id

  const int tiles_per_head = T_ / 16;           // 128
  const int bh = wave / tiles_per_head;          // 0..23
  const int t0 = (wave % tiles_per_head) * 16;   // query tile start

  // ---- Q^T in B-layout: lane=query col, K(=head dim) split by half-wave ----
  const float* qrow = q + ((size_t)bh * T_ + (t0 + lq)) * D_;
  V16 qb0, qb1;
#pragma unroll
  for (int j = 0; j < 16; ++j) {
    qb0.h[j] = (_Float16)qrow[hf * 8 + (j & 7) + (j >> 3) * 16];        // dims 0..31
    qb1.h[j] = (_Float16)qrow[32 + hf * 8 + (j & 7) + (j >> 3) * 16];   // dims 32..63
  }

  v8f acc[4];
#pragma unroll
  for (int i = 0; i < 4; ++i) acc[i] = 0.f;
  float m_run = -__builtin_inff();
  float l_run = 0.f;
  const float scale = 0.125f;                      // 1/sqrt(64)

  const _Float16* kbase  = k16  + (size_t)bh * T_ * D_;
  const _Float16* vtbase = vt16 + (size_t)bh * D_ * T_;
  const float*   biasrow = bias + ((size_t)bh * T_ + (t0 + lq)) * T_;

  const int nblocks = (t0 + 16 + 31) / 32;
  for (int blk = 0; blk < nblocks; ++blk) {
    const int s0 = blk * 32;
    // prefetch next block's bias row
    if (blk + 1 < nblocks)
      __builtin_prefetch(biasrow + s0 + 32, 0, 3);

    // ---- S^T tiles: c0 = keys s0..s0+15, c1 = keys s0+16..s0+31 ----
    const _Float16* kr0 = kbase + (size_t)(s0 + lq) * D_;
    const _Float16* kr1 = kbase + (size_t)(s0 + 16 + lq) * D_;
    V16 a00, a01, a10, a11;
    a00.q[0] = *(const uint4*)(kr0 + hf * 8);        // dims 0..31 tile
    a00.q[1] = *(const uint4*)(kr0 + 16 + hf * 8);
    a01.q[0] = *(const uint4*)(kr0 + 32 + hf * 8);   // dims 32..63 tile
    a01.q[1] = *(const uint4*)(kr0 + 48 + hf * 8);
    a10.q[0] = *(const uint4*)(kr1 + hf * 8);
    a10.q[1] = *(const uint4*)(kr1 + 16 + hf * 8);
    a11.q[0] = *(const uint4*)(kr1 + 32 + hf * 8);
    a11.q[1] = *(const uint4*)(kr1 + 48 + hf * 8);

    v8f c0 = 0.f, c1 = 0.f;
    c0 = __builtin_amdgcn_wmma_f32_16x16x32_f16(false, a00.v, false, qb0.v, (short)0, c0, false, false);
    c0 = __builtin_amdgcn_wmma_f32_16x16x32_f16(false, a01.v, false, qb1.v, (short)0, c0, false, false);
    c1 = __builtin_amdgcn_wmma_f32_16x16x32_f16(false, a10.v, false, qb0.v, (short)0, c1, false, false);
    c1 = __builtin_amdgcn_wmma_f32_16x16x32_f16(false, a11.v, false, qb1.v, (short)0, c1, false, false);

    // ---- scale + bias + causal mask ----
    F8 sA; sA.v = c0;    // key = s0 + hf*8 + r
    F8 sB; sB.v = c1;    // key = s0 + 16 + hf*8 + r
    float bv0[8], bv1[8];
#pragma unroll
    for (int r = 0; r < 8; ++r) {
      bv0[r] = biasrow[s0 + hf * 8 + r];
      bv1[r] = biasrow[s0 + 16 + hf * 8 + r];
    }
    const int qidx = t0 + lq;
    const bool domask = (s0 + 31 > t0);
#pragma unroll
    for (int r = 0; r < 8; ++r) {
      const int key0 = s0 + hf * 8 + r;
      const int key1 = key0 + 16;
      float v0 = sA.f[r] * scale + bv0[r];
      float v1 = sB.f[r] * scale + bv1[r];
      if (domask) {
        if (key0 > qidx) v0 = -__builtin_inff();
        if (key1 > qidx) v1 = -__builtin_inff();
      }
      sA.f[r] = v0; sB.f[r] = v1;
    }

    // ---- online softmax (per-query = per-lane-pair state) ----
    float mloc = -__builtin_inff();
#pragma unroll
    for (int r = 0; r < 8; ++r)
      mloc = fmaxf(mloc, fmaxf(sA.f[r], sB.f[r]));
    mloc = fmaxf(mloc, shfl16f(mloc));
    const float m_new = fmaxf(m_run, mloc);
    const float alpha = __expf(m_run - m_new);

    H8 plo, phi;
    float psum = 0.f;
#pragma unroll
    for (int r = 0; r < 8; ++r) {
      float p0 = __expf(sA.f[r] - m_new);
      float p1 = __expf(sB.f[r] - m_new);
      psum += p0 + p1;
      plo.h[r] = (_Float16)p0;
      phi.h[r] = (_Float16)p1;
    }
    psum += shfl16f(psum);
    l_run = l_run * alpha + psum;
    m_run = m_new;
#pragma unroll
    for (int i = 0; i < 4; ++i) acc[i] = acc[i] * alpha;

    // ---- P^T -> B-layout via one half-wave exchange ----
    H8 olo, ohi;
#pragma unroll
    for (int i = 0; i < 4; ++i) {
      olo.u[i] = shfl16u(plo.u[i]);
      ohi.u[i] = shfl16u(phi.u[i]);
    }
    V16 pb;
#pragma unroll
    for (int j = 0; j < 8; ++j) {
      pb.h[j]     = (hf == 0) ? plo.h[j] : ohi.h[j];  // K = keys 0..7 / 16..23
      pb.h[8 + j] = (hf == 0) ? olo.h[j] : phi.h[j];  // K = keys 8..15 / 24..31
    }

    // ---- y^T += V^T P^T over 4 D-tiles ----
#pragma unroll
    for (int dt = 0; dt < 4; ++dt) {
      const _Float16* vr = vtbase + (size_t)(dt * 16 + lq) * T_ + s0;
      V16 va;
      va.q[0] = *(const uint4*)(vr + hf * 16);
      va.q[1] = *(const uint4*)(vr + hf * 16 + 8);
      acc[dt] = __builtin_amdgcn_wmma_f32_16x16x32_f16(false, va.v, false, pb.v, (short)0, acc[dt], false, false);
    }
  }

  // ---- finalize: y[b, t0+lq, h*64 + d] = acc/l ----
  const float inv_l = 1.f / l_run;
  const int b_ = bh / H_;
  const int hh = bh % H_;
#pragma unroll
  for (int dt = 0; dt < 4; ++dt) {
    F8 o; o.v = acc[dt];
    H8 oh;
#pragma unroll
    for (int r = 0; r < 8; ++r) oh.h[r] = (_Float16)(o.f[r] * inv_l);
    _Float16* dst = y16 + ((size_t)b_ * T_ + t0 + lq) * DIM_
                        + hh * 64 + dt * 16 + hf * 8;
    *(uint4*)dst = oh.q;
  }
}

// ---------------------------------------------------------------------------
// Projection: out(4096 x 768) = y16 @ W ; uses W^T f16 so B loads contiguous.
// 1 wave = 16 rows x 64 cols output tile.
// ---------------------------------------------------------------------------
__global__ __launch_bounds__(128) void proj_kernel(
    const _Float16* __restrict__ y16, const _Float16* __restrict__ wt16,
    float* __restrict__ out) {
  const int wave = blockIdx.x * (blockDim.x >> 5) + (threadIdx.x >> 5);
  const int lane = threadIdx.x & 31;
  const int lq = lane & 15, hf = lane >> 4;
  const int ntiles = DIM_ / 64;                  // 12
  const int mtile = wave / ntiles;               // 0..255
  const int n0 = (wave % ntiles) * 64;
  const int row0 = mtile * 16;

  v8f c[4];
#pragma unroll
  for (int i = 0; i < 4; ++i) c[i] = 0.f;

  for (int k0 = 0; k0 < DIM_; k0 += 32) {
    const _Float16* arow = y16 + (size_t)(row0 + lq) * DIM_ + k0;
    V16 a;
    a.q[0] = *(const uint4*)(arow + hf * 16);
    a.q[1] = *(const uint4*)(arow + hf * 16 + 8);
#pragma unroll
    for (int s = 0; s < 4; ++s) {
      const _Float16* brow = wt16 + (size_t)(n0 + s * 16 + lq) * DIM_ + k0 + hf * 16;
      V16 bm;
      bm.q[0] = *(const uint4*)(brow);
      bm.q[1] = *(const uint4*)(brow + 8);
      c[s] = __builtin_amdgcn_wmma_f32_16x16x32_f16(false, a.v, false, bm.v, (short)0, c[s], false, false);
    }
  }

#pragma unroll
  for (int s = 0; s < 4; ++s) {
    F8 o; o.v = c[s];
#pragma unroll
    for (int r = 0; r < 8; ++r) {
      const int row = row0 + hf * 8 + r;
      const int col = n0 + s * 16 + lq;
      out[(size_t)row * DIM_ + col] = o.f[r];
    }
  }
}

// ---------------------------------------------------------------------------
extern "C" void kernel_launch(void* const* d_in, const int* in_sizes, int n_in,
                              void* d_out, int out_size, void* d_ws, size_t ws_size,
                              hipStream_t stream) {
  const float* q    = (const float*)d_in[0];
  const float* k    = (const float*)d_in[1];
  const float* v    = (const float*)d_in[2];
  const float* bias = (const float*)d_in[3];
  const float* W    = (const float*)d_in[4];
  float* out = (float*)d_out;

  const size_t NKV = (size_t)B_ * H_ * T_ * D_;      // 3,145,728
  _Float16* k16  = (_Float16*)d_ws;
  _Float16* vt16 = k16 + NKV;
  _Float16* wt16 = vt16 + NKV;
  _Float16* y16  = wt16 + (size_t)DIM_ * DIM_;
  // total ws use: (3*NKV + DIM*DIM) * 2 bytes ~= 20.1 MB

  prep_kernel<<<2048, 256, 0, stream>>>(k, v, W, k16, vt16, wt16);

  // waves = B*H*(T/16) = 3072; 4 waves/block
  attn_kernel<<<768, 128, 0, stream>>>(q, k16, vt16, bias, y16);

  // waves = (B*T/16) * (DIM/64) = 256*12 = 3072; 4 waves/block
  proj_kernel<<<768, 128, 0, stream>>>(y16, wt16, out);
}
